// SGC_20761871909284
// MI455X (gfx1250) — matrix operand order repaired and checked by reading, not compiled, not run on hardware
//
#include <hip/hip_runtime.h>
#include <stdint.h>

// MI455X / gfx1250 implementation.
// Pipeline:
//   1) convert adj (f32)  -> W0 (bf16)          [elementwise]
//   2) convert rm  (f32)  -> RMB (bf16)         [elementwise]
//   3) W1 = W0 @ W0   (adj^2)                    bf16 WMMA, f32 accum
//   4) W0 = (W1 @ W1) with zeroed diagonal      (adj^4 masked)
//   5) out[b] = W0 @ RMB[b]  (4096x4096 x 4096x32, fp32 out)
// Workspace: 32MB + 32MB + 2MB = 66MB required in d_ws.
//
// GEMM: block tile 256x128, K-step 32, 8 waves (4x2), wave tile 64x64
// (16 WMMA frags). A tile fetched by the Tensor Data Mover into LDS with
// pad (4 dwords per 16 dwords -> stride 40 bf16, conflict-free b128 reads).
// B tile transposed into LDS by the threads. Double-buffered LDS, one
// barrier per K-step; tile k+1 prefetched while computing tile k.

typedef __bf16 bf16_t;
typedef __attribute__((ext_vector_type(16))) __bf16 v16bf;
typedef __attribute__((ext_vector_type(8)))  float  v8f;
typedef __attribute__((ext_vector_type(4)))  unsigned int v4u;
typedef __attribute__((ext_vector_type(8)))  int v8i;
typedef __attribute__((ext_vector_type(4)))  int v4i;

// named punning unions (assignable)
union FragU  { float4 f[2]; v16bf v; bf16_t h[16]; };
union Rm4U   { float2 f2; bf16_t h[4]; };

#define N4K 4096
#define BT_STRIDE 40   // bf16 elements per LDS row (32 data + 8 pad) -> 80B, 16B aligned
#define KSTEPS (N4K / 32)

// ---------------------------------------------------------------------------
// f32 -> bf16 conversion, 4 elements / thread
// ---------------------------------------------------------------------------
__global__ __launch_bounds__(256)
void f32_to_bf16_kernel(const float* __restrict__ in, bf16_t* __restrict__ out, int n4) {
    int i = blockIdx.x * blockDim.x + threadIdx.x;
    if (i < n4) {
        float4 v = ((const float4*)in)[i];
        union { bf16_t h[4]; float2 f2; } u;
        u.h[0] = (bf16_t)v.x; u.h[1] = (bf16_t)v.y;
        u.h[2] = (bf16_t)v.z; u.h[3] = (bf16_t)v.w;
        ((float2*)out)[i] = u.f2;
    }
}

// ---------------------------------------------------------------------------
// TDM load of a rows x 32 bf16 tile (row-major, row stride N4K elems) into
// LDS at lds_off, padded 4 dwords every 16 dwords -> LDS row stride 40 bf16.
// ---------------------------------------------------------------------------
__device__ __forceinline__
void tdm_load_tileA(uint32_t lds_off, const bf16_t* gptr, int rows) {
    uint64_t ga = (uint64_t)(uintptr_t)gptr;
    v4u g0 = { 1u,                                     // count=1, user descriptor
               lds_off,                                // lds_addr (bytes)
               (unsigned)(ga & 0xFFFFFFFFull),
               (unsigned)((ga >> 32) & 0x01FFFFFFull) | (2u << 30) }; // type=2 (image)
    // data_size=1 (2B) | pad_enable | pad_interval=3 (16 dwords) | pad_amount=3 (4 dwords)
    v8i g1 = { (int)0x06D10000,
               (int)0x10000000,        // tensor_dim0 = 4096 (low16 in [31:16])
               (int)0x10000000,        // tensor_dim1 = 4096 (low16 in [31:16])
               (int)(32 << 16),        // tile_dim0 = 32 (K, contiguous)
               rows,                   // tile_dim1 = rows
               N4K,                    // tensor_dim0_stride = 4096 (low 32)
               0, 0 };
    v4i gz = { 0, 0, 0, 0 };
#if defined(__clang_major__) && (__clang_major__ >= 23)
    v8i gz8 = { 0, 0, 0, 0, 0, 0, 0, 0 };
    __builtin_amdgcn_tensor_load_to_lds(g0, g1, gz, gz, gz8, 0);
#else
    __builtin_amdgcn_tensor_load_to_lds(g0, g1, gz, gz, 0);
#endif
}

// ---------------------------------------------------------------------------
// C = A @ A  (N4K x N4K), bf16 in, bf16 out, fp32 accumulate.
// Block = 256 threads (8 waves), tile 256(M) x 128(N), K-step 32.
// Waves: 4 (M) x 2 (N); wave tile 64x64 = 4x4 WMMA fragments.
// ---------------------------------------------------------------------------
template<bool MASK_DIAG>
__global__ __launch_bounds__(256)
void gemm_sq_kernel(const bf16_t* __restrict__ A, bf16_t* __restrict__ C) {
    __shared__ bf16_t lsA[2][256 * BT_STRIDE];   // [m][k], stride 40 (TDM pad)
    __shared__ bf16_t lsB[2][128 * BT_STRIDE];   // [n][k] transposed, stride 40

    const int tid  = threadIdx.x;
    const int hf   = (tid >> 4) & 1;          // lane half
    const int l15  = tid & 15;
    const int w    = tid >> 5;                // wave id 0..7
    const int wm   = w >> 1;                  // 0..3 -> 64 rows each
    const int wn   = w & 1;                   // 0..1 -> 64 cols each
    const int mTile = blockIdx.y * 256;
    const int nTile = blockIdx.x * 128;

    // B transpose-loader assignment: thread -> (k, 16-col group)
    const int bk  = tid & 31;
    const int bn0 = (tid >> 5) * 16;

    v8f acc[4][4];
#pragma unroll
    for (int mi = 0; mi < 4; ++mi)
#pragma unroll
        for (int ni = 0; ni < 4; ++ni) acc[mi][ni] = (v8f)0.0f;

    const uint32_t ldsA_off[2] = { (uint32_t)(uintptr_t)(&lsA[0][0]),
                                   (uint32_t)(uintptr_t)(&lsA[1][0]) };

    // ---- prologue: stage tile 0 into buffer 0 ----
    if (tid < 32) tdm_load_tileA(ldsA_off[0], A + (size_t)mTile * N4K, 256);
    {
        const bf16_t* src = A + (size_t)bk * N4K + (nTile + bn0);
        FragU u;
        u.f[0] = ((const float4*)src)[0];
        u.f[1] = ((const float4*)src)[1];
#pragma unroll
        for (int i = 0; i < 16; ++i)
            lsB[0][(bn0 + i) * BT_STRIDE + bk] = u.h[i];
    }
    if (tid < 32) __builtin_amdgcn_s_wait_tensorcnt(0);
    __syncthreads();

    for (int it = 0; it < KSTEPS; ++it) {
        const int cur = it & 1, nxt = cur ^ 1;
        const bool hasNext = (it + 1) < KSTEPS;

        // ---- prefetch tile it+1 (TDM -> lsA[nxt]; global B -> regs) ----
        FragU bPre;
        if (hasNext) {
            if (tid < 32)
                tdm_load_tileA(ldsA_off[nxt],
                               A + (size_t)mTile * N4K + (it + 1) * 32, 256);
            const bf16_t* src = A + (size_t)((it + 1) * 32 + bk) * N4K + (nTile + bn0);
            bPre.f[0] = ((const float4*)src)[0];
            bPre.f[1] = ((const float4*)src)[1];
        }

        // ---- compute on buffer cur: fragments per ISA 7.12.2 layouts ----
        v16bf af[4], bfr[4];
#pragma unroll
        for (int mi = 0; mi < 4; ++mi) {
            const bf16_t* pr = &lsA[cur][(wm * 64 + mi * 16 + l15) * BT_STRIDE];
            FragU ua;
            ua.f[0] = *(const float4*)(pr + hf * 8);        // K 0-7 / 8-15
            ua.f[1] = *(const float4*)(pr + 16 + hf * 8);   // K 16-23 / 24-31
            af[mi] = ua.v;
        }
#pragma unroll
        for (int ni = 0; ni < 4; ++ni) {
            const bf16_t* pr = &lsB[cur][(wn * 64 + ni * 16 + l15) * BT_STRIDE];
            FragU ub;
            ub.f[0] = *(const float4*)(pr + hf * 8);        // K 0-7 / 8-15
            ub.f[1] = *(const float4*)(pr + 16 + hf * 8);   // K 16-23 / 24-31
            bfr[ni] = ub.v;
        }
#pragma unroll
        for (int mi = 0; mi < 4; ++mi)
#pragma unroll
            for (int ni = 0; ni < 4; ++ni)
                acc[mi][ni] = __builtin_amdgcn_wmma_f32_16x16x32_bf16(
                    false, af[mi], false, bfr[ni], (short)0, acc[mi][ni], false, false);

        // ---- publish tile it+1 ----
        if (hasNext) {
#pragma unroll
            for (int i = 0; i < 16; ++i)
                lsB[nxt][(bn0 + i) * BT_STRIDE + bk] = bPre.h[i];
            if (tid < 32) __builtin_amdgcn_s_wait_tensorcnt(0);
        }
        __syncthreads();
    }

    // ---- store bf16, optional diagonal mask ----
#pragma unroll
    for (int mi = 0; mi < 4; ++mi)
#pragma unroll
        for (int ni = 0; ni < 4; ++ni) {
            const int col  = nTile + wn * 64 + ni * 16 + l15;
            const int rowb = mTile + wm * 64 + mi * 16 + hf * 8;
#pragma unroll
            for (int v = 0; v < 8; ++v) {
                const int row = rowb + v;
                float x = acc[mi][ni][v];
                if (MASK_DIAG && row == col) x = 0.0f;
                C[(size_t)row * N4K + col] = (bf16_t)x;
            }
        }
}

// ---------------------------------------------------------------------------
// out[b,i,f] = sum_j adj4m[i,j] * rm_bf16[b,j,f]   (M=4096, N=32, K=4096)
// Grid: (4096/128, 8). Wave w -> rows [bx*128 + w*16, +16), 2 N-fragments.
// rm tile (32x32) double-buffered transposed in LDS; next A fragment
// prefetched into registers during the WMMAs; one barrier per K-step.
// ---------------------------------------------------------------------------
__global__ __launch_bounds__(256)
void out_gemm_kernel(const bf16_t* __restrict__ A4, const bf16_t* __restrict__ RMB,
                     float* __restrict__ out) {
    __shared__ bf16_t lsR[2][32 * BT_STRIDE];    // [f][k]

    const int tid  = threadIdx.x;
    const int hf   = (tid >> 4) & 1;
    const int l15  = tid & 15;
    const int w    = tid >> 5;
    const int b    = blockIdx.y;
    const int iBase = blockIdx.x * 128 + w * 16;
    const bf16_t* rmB = RMB + (size_t)b * N4K * 32;
    const bf16_t* ar  = A4 + (size_t)(iBase + l15) * N4K;   // this lane's A row

    const int lk  = tid & 31;                 // rm loader: k within tile
    const int lf0 = (tid >> 5) * 4;           // rm loader: 4 consecutive f

    v8f acc[2];
    acc[0] = (v8f)0.0f; acc[1] = (v8f)0.0f;

    // ---- prologue: stage rm tile 0; prefetch A fragment 0 ----
    {
        Rm4U u;
        u.f2 = *(const float2*)(rmB + (size_t)lk * 32 + lf0);
#pragma unroll
        for (int i = 0; i < 4; ++i)
            lsR[0][(lf0 + i) * BT_STRIDE + lk] = u.h[i];
    }
    FragU ua;
    ua.f[0] = *(const float4*)(ar + hf * 8);
    ua.f[1] = *(const float4*)(ar + 16 + hf * 8);
    __syncthreads();

    for (int it = 0; it < KSTEPS; ++it) {
        const int cur = it & 1, nxt = cur ^ 1;
        const bool hasNext = (it + 1) < KSTEPS;

        // prefetch next rm tile + next A fragment into registers
        Rm4U rn;
        FragU un;
        if (hasNext) {
            rn.f2 = *(const float2*)(rmB + (size_t)((it + 1) * 32 + lk) * 32 + lf0);
            const bf16_t* an = ar + (it + 1) * 32;
            un.f[0] = *(const float4*)(an + hf * 8);
            un.f[1] = *(const float4*)(an + 16 + hf * 8);
        }

        // compute on buffer cur
        v16bf af = ua.v;
#pragma unroll
        for (int ni = 0; ni < 2; ++ni) {
            const bf16_t* pr = &lsR[cur][(ni * 16 + l15) * BT_STRIDE];
            FragU ub;
            ub.f[0] = *(const float4*)(pr + hf * 8);
            ub.f[1] = *(const float4*)(pr + 16 + hf * 8);
            acc[ni] = __builtin_amdgcn_wmma_f32_16x16x32_bf16(
                false, af, false, ub.v, (short)0, acc[ni], false, false);
        }

        // publish next rm tile
        if (hasNext) {
#pragma unroll
            for (int i = 0; i < 4; ++i)
                lsR[nxt][(lf0 + i) * BT_STRIDE + lk] = rn.h[i];
            ua = un;
        }
        __syncthreads();
    }

    float* ob = out + (size_t)b * N4K * 32;
#pragma unroll
    for (int ni = 0; ni < 2; ++ni) {
        const int col  = ni * 16 + l15;
        const int rowb = iBase + hf * 8;
#pragma unroll
        for (int v = 0; v < 8; ++v)
            ob[(size_t)(rowb + v) * 32 + col] = acc[ni][v];
    }
}

// ---------------------------------------------------------------------------
// Launcher. Inputs: d_in[0]=regional_means (8,4096,32) f32; d_in[1]=adj (4096^2) f32.
// Output: (8,4096,32) f32. Workspace: needs >= 66 MB.
// ---------------------------------------------------------------------------
extern "C" void kernel_launch(void* const* d_in, const int* in_sizes, int n_in,
                              void* d_out, int out_size, void* d_ws, size_t ws_size,
                              hipStream_t stream) {
    (void)in_sizes; (void)n_in; (void)out_size; (void)ws_size;
    const float* rm  = (const float*)d_in[0];
    const float* adj = (const float*)d_in[1];
    float* out = (float*)d_out;

    const size_t MATB = (size_t)N4K * N4K * sizeof(bf16_t);   // 32 MB
    bf16_t* W0  = (bf16_t*)d_ws;
    bf16_t* W1  = (bf16_t*)((char*)d_ws + MATB);
    bf16_t* RMB = (bf16_t*)((char*)d_ws + 2 * MATB);

    const int nAdj4 = (N4K * N4K) / 4;           // 4,194,304
    const int nRm4  = (8 * N4K * 32) / 4;        // 262,144
    f32_to_bf16_kernel<<<(nAdj4 + 255) / 256, 256, 0, stream>>>(adj, W0, nAdj4);
    f32_to_bf16_kernel<<<(nRm4 + 255) / 256, 256, 0, stream>>>(rm, RMB, nRm4);

    dim3 g(N4K / 128, N4K / 256);                // 32 x 16 workgroups
    gemm_sq_kernel<false><<<g, 256, 0, stream>>>(W0, W1);   // adj^2
    gemm_sq_kernel<true ><<<g, 256, 0, stream>>>(W1, W0);   // adj^4, diag=0

    out_gemm_kernel<<<dim3(N4K / 128, 8), 256, 0, stream>>>(W0, RMB, out);
}
